// Model_41807211659928
// MI455X (gfx1250) — matrix operand order repaired and checked by reading, not compiled, not run on hardware
//
#include <hip/hip_runtime.h>

typedef __attribute__((ext_vector_type(16))) _Float16 v16h;
typedef __attribute__((ext_vector_type(2)))  _Float16 v2h;
typedef __attribute__((ext_vector_type(8)))  float    v8f;
typedef __attribute__((ext_vector_type(4)))  float    v4f;

#define RNN_SEQ 4096
#define RNN_B   256
#define RNN_H   64

union HPack { v2h h; int i; };
union Frag  { int i[8]; v16h v; };

__device__ __forceinline__ float tanh_fast(float x) {
#if __has_builtin(__builtin_amdgcn_tanhf)
  return __builtin_amdgcn_tanhf(x);              // v_tanh_f32 (CDNA5 trans op)
#elif __has_builtin(__builtin_amdgcn_exp2f)
  float e = __builtin_amdgcn_exp2f(2.8853900817779268f * x);  // 2x*log2(e)
  return 1.0f - 2.0f / (e + 1.0f);
#else
  float e = __builtin_exp2f(2.8853900817779268f * x);
  return 1.0f - 2.0f / (e + 1.0f);
#endif
}

__device__ __forceinline__ v8f wmma16(v16h a, v16h b, v8f c) {
  return __builtin_amdgcn_wmma_f32_16x16x32_f16(false, a, false, b, (short)0, c,
                                                false, false);
}

// Pack two f32 -> one VGPR of 2x f16 (lowers to v_cvt_pk_f16_f32).
__device__ __forceinline__ int pack2(float a, float b) {
  HPack p;
  p.h[0] = (_Float16)a;
  p.h[1] = (_Float16)b;
  return p.i;
}

// Standard A-fragment (16x32, f16) of a row-major f32 matrix W[64][64].
// Used for W_hh (pairs with the D-packed h B-fragments).
__device__ __forceinline__ v16h load_a_std(const float* __restrict__ W, int jt,
                                           int ks, int lane) {
  int m  = lane & 15;
  int hi = lane >> 4;
  const float* p = W + (jt * 16 + m) * 64 + ks + hi * 8;
  v4f r0 = ((const v4f*)p)[0];
  v4f r1 = ((const v4f*)p)[1];
  v4f r2 = ((const v4f*)(p + 16))[0];
  v4f r3 = ((const v4f*)(p + 16))[1];
  v16h f;
  f[0]  = (_Float16)r0.x; f[1]  = (_Float16)r0.y; f[2]  = (_Float16)r0.z; f[3]  = (_Float16)r0.w;
  f[4]  = (_Float16)r1.x; f[5]  = (_Float16)r1.y; f[6]  = (_Float16)r1.z; f[7]  = (_Float16)r1.w;
  f[8]  = (_Float16)r2.x; f[9]  = (_Float16)r2.y; f[10] = (_Float16)r2.z; f[11] = (_Float16)r2.w;
  f[12] = (_Float16)r3.x; f[13] = (_Float16)r3.y; f[14] = (_Float16)r3.z; f[15] = (_Float16)r3.w;
  return f;
}

// K-PERMUTED A-fragment for W_ih.  The x B-fragments are loaded as
// 16 contiguous f32 per lane, realizing K permutation
//   hwK 0-7 -> h 0-7, hwK 8-15 -> h 16-23, hwK 16-23 -> h 8-15, hwK 24-31 -> h 24-31.
// Loading W_ih with the same permutation keeps the product exact; each lane reads
// 16 contiguous floats.
__device__ __forceinline__ v16h load_a_perm(const float* __restrict__ W, int jt,
                                            int ks, int lane) {
  int m  = lane & 15;
  int hi = lane >> 4;
  const float* p = W + (jt * 16 + m) * 64 + ks + hi * 16;
  v4f r0 = ((const v4f*)p)[0];
  v4f r1 = ((const v4f*)p)[1];
  v4f r2 = ((const v4f*)p)[2];
  v4f r3 = ((const v4f*)p)[3];
  v16h f;
  f[0]  = (_Float16)r0.x; f[1]  = (_Float16)r0.y; f[2]  = (_Float16)r0.z; f[3]  = (_Float16)r0.w;
  f[4]  = (_Float16)r1.x; f[5]  = (_Float16)r1.y; f[6]  = (_Float16)r1.z; f[7]  = (_Float16)r1.w;
  f[8]  = (_Float16)r2.x; f[9]  = (_Float16)r2.y; f[10] = (_Float16)r2.z; f[11] = (_Float16)r2.w;
  f[12] = (_Float16)r3.x; f[13] = (_Float16)r3.y; f[14] = (_Float16)r3.z; f[15] = (_Float16)r3.w;
  return f;
}

// Non-temporal load of an x tile [16 rows x 64 cols] (f32): raw v4f's for two
// K-permuted B-fragments (per lane: 16 contiguous f32 per fragment).
__device__ __forceinline__ void load_x(const float* __restrict__ xrow, int lane,
                                       v4f r[8]) {
  int n  = lane & 15;
  int hi = lane >> 4;
  const v4f* p = (const v4f*)(xrow + n * 64 + hi * 16);
  r[0] = __builtin_nontemporal_load(p + 0);
  r[1] = __builtin_nontemporal_load(p + 1);
  r[2] = __builtin_nontemporal_load(p + 2);
  r[3] = __builtin_nontemporal_load(p + 3);
  const v4f* q = p + 8;                  // +32 floats
  r[4] = __builtin_nontemporal_load(q + 0);
  r[5] = __builtin_nontemporal_load(q + 1);
  r[6] = __builtin_nontemporal_load(q + 2);
  r[7] = __builtin_nontemporal_load(q + 3);
}

__device__ __forceinline__ v16h cvt_frag(const v4f* r) {
  v16h f;
  f[0]  = (_Float16)r[0].x; f[1]  = (_Float16)r[0].y; f[2]  = (_Float16)r[0].z; f[3]  = (_Float16)r[0].w;
  f[4]  = (_Float16)r[1].x; f[5]  = (_Float16)r[1].y; f[6]  = (_Float16)r[1].z; f[7]  = (_Float16)r[1].w;
  f[8]  = (_Float16)r[2].x; f[9]  = (_Float16)r[2].y; f[10] = (_Float16)r[2].z; f[11] = (_Float16)r[2].w;
  f[12] = (_Float16)r[3].x; f[13] = (_Float16)r[3].y; f[14] = (_Float16)r[3].z; f[15] = (_Float16)r[3].w;
  return f;
}

// One RNN timestep for one wave (16 batch rows).
// buf holds x(t+1) on entry; it is consumed and refilled with x(t+3).
__device__ __forceinline__ void rnn_step(
    int t, v4f buf[8], const v16h Aih[4][2], const v16h Ahh[4][2],
    const v8f cb[4], v8f cx[4], v16h& Bh0, v16h& Bh1,
    const float* __restrict__ x, float* __restrict__ out, int b0, int n, int hi,
    int lane) {
  // ---- 1) critical path: recurrent GEMM (2 chained WMMAs per tile) ----
  v8f d[4];
#pragma unroll
  for (int jt = 0; jt < 4; ++jt) {
    v8f c = wmma16(Ahh[jt][0], Bh0, cx[jt]);
    d[jt] = wmma16(Ahh[jt][1], Bh1, c);
  }

  // ---- 2) tanh (independent across all 32 values) ----
#pragma unroll
  for (int jt = 0; jt < 4; ++jt)
#pragma unroll
    for (int r = 0; r < 8; ++r) d[jt][r] = tanh_fast(d[jt][r]);

  // ---- 3) D -> next-step B layout: straight packs (table-order Ahh) ----
  Frag f0, f1;
#pragma unroll
  for (int i = 0; i < 4; ++i) {
    f0.i[i]     = pack2(d[0][2 * i], d[0][2 * i + 1]);
    f0.i[4 + i] = pack2(d[1][2 * i], d[1][2 * i + 1]);
    f1.i[i]     = pack2(d[2][2 * i], d[2][2 * i + 1]);
    f1.i[4 + i] = pack2(d[3][2 * i], d[3][2 * i + 1]);
  }
  Bh0 = f0.v;
  Bh1 = f1.v;

  // ---- 4) off critical path: next step's input projection ----
  {
    v16h Bx0 = cvt_frag(&buf[0]);
    v16h Bx1 = cvt_frag(&buf[4]);
#pragma unroll
    for (int jt = 0; jt < 4; ++jt) {
      v8f c  = wmma16(Aih[jt][0], Bx0, cb[jt]);
      cx[jt] = wmma16(Aih[jt][1], Bx1, c);
    }
  }

  // ---- 5) refill this buffer with x(t+3)  (prefetch distance = 2 steps) ----
  int tpre = (t + 3 < RNN_SEQ) ? (t + 3) : (RNN_SEQ - 1);
  load_x(x + (size_t)tpre * (RNN_B * RNN_H) + (size_t)b0 * RNN_H, lane, buf);

  // ---- 6) store h_t (f32, non-temporal): per lane 8-float contiguous runs ----
  float* orow = out + ((size_t)t * RNN_B + b0 + n) * RNN_H + hi * 8;
#pragma unroll
  for (int jt = 0; jt < 4; ++jt) {
    v4f v0 = { d[jt][0], d[jt][1], d[jt][2], d[jt][3] };
    v4f v1 = { d[jt][4], d[jt][5], d[jt][6], d[jt][7] };
    __builtin_nontemporal_store(v0, ((v4f*)(orow + jt * 16)) + 0);
    __builtin_nontemporal_store(v1, ((v4f*)(orow + jt * 16)) + 1);
  }
}

// One wave (32 lanes) per 16 batch rows: h_t^T = tanh(Wih x^T + b + Whh h^T)
// as 4 D-tiles [16(j) x 16(b)] per step, entirely in registers. 16 blocks total.
__global__ __launch_bounds__(32, 1)
void rnn_scan_kernel(const float* __restrict__ x, const float* __restrict__ w_ih,
                     const float* __restrict__ w_hh, const float* __restrict__ b_ih,
                     const float* __restrict__ b_hh, float* __restrict__ out) {
  const int lane = threadIdx.x & 31;
  const int b0   = blockIdx.x * 16;
  const int n    = lane & 15;
  const int hi   = lane >> 4;

  // Resident weight A-fragments (one-time load, ~128 VGPRs).
  v16h Aih[4][2], Ahh[4][2];
#pragma unroll
  for (int jt = 0; jt < 4; ++jt) {
    Aih[jt][0] = load_a_perm(w_ih, jt, 0, lane);
    Aih[jt][1] = load_a_perm(w_ih, jt, 32, lane);
    Ahh[jt][0] = load_a_std(w_hh, jt, 0, lane);     // table K order (matches D-pack B)
    Ahh[jt][1] = load_a_std(w_hh, jt, 32, lane);
  }

  // Bias as C-matrix init: D layout lane holds rows j = jt*16 + hi*8 + r.
  v8f cb[4];
#pragma unroll
  for (int jt = 0; jt < 4; ++jt) {
    const float* p1 = b_ih + jt * 16 + hi * 8;
    const float* p2 = b_hh + jt * 16 + hi * 8;
#pragma unroll
    for (int r = 0; r < 8; ++r) cb[jt][r] = p1[r] + p2[r];
  }

  // h_0 = 0
  v16h Bh0, Bh1;
#pragma unroll
  for (int i = 0; i < 16; ++i) { Bh0[i] = (_Float16)0.0f; Bh1[i] = (_Float16)0.0f; }

  // Prologue: cx for t=0 from x(0); then prefetch x(1) and x(2) (ping-pong).
  v8f cx[4];
  v4f xrA[8], xrB[8];
  load_x(x + (size_t)b0 * RNN_H, lane, xrA);
  {
    v16h Bx0 = cvt_frag(&xrA[0]);
    v16h Bx1 = cvt_frag(&xrA[4]);
#pragma unroll
    for (int jt = 0; jt < 4; ++jt) {
      v8f c  = wmma16(Aih[jt][0], Bx0, cb[jt]);
      cx[jt] = wmma16(Aih[jt][1], Bx1, c);
    }
  }
  load_x(x + (size_t)1 * (RNN_B * RNN_H) + (size_t)b0 * RNN_H, lane, xrA);
  load_x(x + (size_t)2 * (RNN_B * RNN_H) + (size_t)b0 * RNN_H, lane, xrB);

#pragma unroll 1
  for (int t = 0; t < RNN_SEQ; t += 2) {
    rnn_step(t,     xrA, Aih, Ahh, cb, cx, Bh0, Bh1, x, out, b0, n, hi, lane);
    rnn_step(t + 1, xrB, Aih, Ahh, cb, cx, Bh0, Bh1, x, out, b0, n, hi, lane);
  }
}

extern "C" void kernel_launch(void* const* d_in, const int* in_sizes, int n_in,
                              void* d_out, int out_size, void* d_ws, size_t ws_size,
                              hipStream_t stream) {
  (void)in_sizes; (void)n_in; (void)out_size; (void)d_ws; (void)ws_size;
  const float* x    = (const float*)d_in[0];
  const float* w_ih = (const float*)d_in[1];
  const float* w_hh = (const float*)d_in[2];
  const float* b_ih = (const float*)d_in[3];
  const float* b_hh = (const float*)d_in[4];
  float* out = (float*)d_out;
  rnn_scan_kernel<<<dim3(RNN_B / 16), dim3(32), 0, stream>>>(x, w_ih, w_hh, b_ih,
                                                             b_hh, out);
}
